// ProbAttention_73100343378035
// MI455X (gfx1250) — compile-verified, matrix operand order checked by reading, not run on hardware
//
#include <hip/hip_runtime.h>
#include <hip/hip_bf16.h>
#include <math.h>

// Problem constants (fixed by the reference)
#define BB 4
#define LL 4096
#define HH 8
#define DD 64
#define UPAD 48   // U=45 padded to 3 WMMA M-tiles

typedef __attribute__((ext_vector_type(16))) _Float16 v16h;
typedef __attribute__((ext_vector_type(8)))  float    v8f;

__device__ __forceinline__ v8f wmma_f16(v16h a, v16h b, v8f c) {
  // probe-confirmed signature: (neg_a, A, neg_b, B, c_mod, C, reuse_a, reuse_b)
  return __builtin_amdgcn_wmma_f32_16x16x32_f16(
      false, a, false, b, (short)0, c, false, false);
}

// ---------------------------------------------------------------------------
// Kernel 1: sparsity measure m(b,h,l) = max_s(q·k_s) - sum_s(q·k_s)/L
// ---------------------------------------------------------------------------
__global__ __launch_bounds__(256)
void compute_m_kernel(const float* __restrict__ q, const float* __restrict__ k,
                      const int* __restrict__ idx, float* __restrict__ mvals, int U) {
  long gid = (long)blockIdx.x * 256 + threadIdx.x;   // < 32*4096
  int bh = (int)(gid >> 12);
  int l  = (int)(gid & 4095);
  int b = bh >> 3, h = bh & 7;

  const float4* qr = (const float4*)(q + (((long)b * LL + l) * HH + h) * DD);
  float4 qv[16];
#pragma unroll
  for (int i = 0; i < 16; ++i) qv[i] = qr[i];

  float mx = -INFINITY, sum = 0.0f;
  for (int s = 0; s < U; ++s) {
    int kl = idx[l * U + s];
    const float4* kr = (const float4*)(k + (((long)b * LL + kl) * HH + h) * DD);
    if (s + 1 < U) {
      int kln = idx[l * U + s + 1];
      __builtin_prefetch(k + (((long)b * LL + kln) * HH + h) * DD, 0, 0);
    }
    float dot = 0.0f;
#pragma unroll
    for (int i = 0; i < 16; ++i) {
      float4 kv = kr[i];
      dot += qv[i].x * kv.x + qv[i].y * kv.y + qv[i].z * kv.z + qv[i].w * kv.w;
    }
    mx = fmaxf(mx, dot);
    sum += dot;
  }
  mvals[gid] = mx - sum * (1.0f / (float)LL);
}

// ---------------------------------------------------------------------------
// Kernel 2: top-U selection per (b,h) via iterative argmax in LDS
// ---------------------------------------------------------------------------
__global__ __launch_bounds__(256)
void topk_kernel(const float* __restrict__ mvals, int* __restrict__ topidx, int U) {
  int bh = blockIdx.x;
  __shared__ float sm[LL];
  __shared__ float rmax[256];
  __shared__ int   rarg[256];
  int tid = threadIdx.x;
  for (int i = tid; i < LL; i += 256) sm[i] = mvals[(long)bh * LL + i];
  __syncthreads();

  for (int it = 0; it < U; ++it) {
    float best = -INFINITY; int barg = 0;
    for (int i = tid; i < LL; i += 256) {
      float x = sm[i];
      if (x > best) { best = x; barg = i; }
    }
    rmax[tid] = best; rarg[tid] = barg;
    __syncthreads();
    for (int s = 128; s > 0; s >>= 1) {
      if (tid < s && rmax[tid + s] > rmax[tid]) {
        rmax[tid] = rmax[tid + s]; rarg[tid] = rarg[tid + s];
      }
      __syncthreads();
    }
    if (tid == 0) {
      topidx[bh * UPAD + it] = rarg[0];
      sm[rarg[0]] = -INFINITY;
    }
    __syncthreads();
  }
  if (tid >= U && tid < UPAD) topidx[bh * UPAD + tid] = 0;
}

// ---------------------------------------------------------------------------
// Kernel 3a: vmean(b,h,d) = mean_l v ; 3b: fill out with broadcast mean
// ---------------------------------------------------------------------------
__global__ __launch_bounds__(256)
void vmean_kernel(const float* __restrict__ v, float* __restrict__ vmean) {
  int bh = blockIdx.x;
  int b = bh >> 3, h = bh & 7;
  int tid = threadIdx.x;
  int d = tid & 63, part = tid >> 6;           // 4 partials per d
  __shared__ float red[256];
  float s = 0.0f;
  for (int l = part; l < LL; l += 4)
    s += v[(((long)b * LL + l) * HH + h) * DD + d];
  red[tid] = s;
  __syncthreads();
  if (part == 0) {
    float t = red[d] + red[64 + d] + red[128 + d] + red[192 + d];
    vmean[bh * DD + d] = t * (1.0f / (float)LL);
  }
}

__global__ __launch_bounds__(256)
void fill_kernel(const float* __restrict__ vmean, float* __restrict__ out) {
  long i = (long)blockIdx.x * 256 + threadIdx.x;   // over B*H*L*D = 2^23
  out[i] = vmean[(int)((i >> 18) << 6) | (int)(i & 63)];
}

// ---------------------------------------------------------------------------
// Fragment loaders (wave32 WMMA layouts, ISA 7.12.2)
// A 16x32 f16: lane m=lane&15, half=lane>>4; VGPR j holds K pair at
//   kpair = (j>>2)*16 + half*8 + (j&3)*2
// ---------------------------------------------------------------------------
__device__ __forceinline__ v16h load_a_qred(const _Float16* sq, int mtile, int kc, int lane) {
  int m = mtile * 16 + (lane & 15);
  int half = lane >> 4;
  v16h a;
#pragma unroll
  for (int j = 0; j < 8; ++j) {
    int kb = kc * 32 + (j >> 2) * 16 + half * 8 + (j & 3) * 2;
    a[2 * j]     = sq[m * DD + kb];
    a[2 * j + 1] = sq[m * DD + kb + 1];
  }
  return a;
}

// B 32x16 = k^T tile: B[kk][n] = keys[b, l=nbase+n, h, d=kc*32+kk]
__device__ __forceinline__ v16h load_b_k(const float* __restrict__ keys,
                                         int b, int h, int nbase, int kc, int lane) {
  int n = lane & 15, half = lane >> 4;
  long rowbase = (((long)b * LL + (nbase + n)) * HH + h) * DD + kc * 32;
  v16h bf;
#pragma unroll
  for (int j = 0; j < 8; ++j) {
    int kb = (j >> 2) * 16 + half * 8 + (j & 3) * 2;
    float2 p = *(const float2*)(keys + rowbase + kb);
    bf[2 * j]     = (_Float16)p.x;
    bf[2 * j + 1] = (_Float16)p.y;
  }
  return bf;
}

// A 16x32 tile of softmaxed P, built on the fly from f32 scores + row stats
__device__ __forceinline__ v16h load_a_p(const float* __restrict__ scores,
                                         const float* smax, const float* sinv,
                                         int bh, int mtile, int kb32, int lane) {
  int u = mtile * 16 + (lane & 15);
  int half = lane >> 4;
  long base = ((long)bh * UPAD + u) * LL + kb32 * 32;
  float rm = smax[u], ri = sinv[u];
  v16h a;
#pragma unroll
  for (int j = 0; j < 8; ++j) {
    int kb = (j >> 2) * 16 + half * 8 + (j & 3) * 2;
    float2 s2 = *(const float2*)(scores + base + kb);
    a[2 * j]     = (_Float16)(__expf(s2.x - rm) * ri);
    a[2 * j + 1] = (_Float16)(__expf(s2.y - rm) * ri);
  }
  return a;
}

// B 32x16 of v: B[kk][n] = v[b, l=kb32*32+kk, h, d=dbase+n]
__device__ __forceinline__ v16h load_b_v(const float* __restrict__ v,
                                         int b, int h, int kb32, int dbase, int lane) {
  int n = lane & 15, half = lane >> 4;
  int d = dbase + n;
  v16h bf;
#pragma unroll
  for (int j = 0; j < 8; ++j) {
    int kk = (j >> 2) * 16 + half * 8 + (j & 3) * 2;
    long l = (long)kb32 * 32 + kk;
    bf[2 * j]     = (_Float16)v[(((long)b * LL + l)     * HH + h) * DD + d];
    bf[2 * j + 1] = (_Float16)v[(((long)b * LL + l + 1) * HH + h) * DD + d];
  }
  return bf;
}

// ---------------------------------------------------------------------------
// Kernel 4: scores[bh, u, l] = scale * q_red[u,:] . k[l,:]   (WMMA)
// grid: (16 n-chunks of 256 cols, 32 bh), block 128 = 4 waves
// ---------------------------------------------------------------------------
__global__ __launch_bounds__(128)
void scores_wmma_kernel(const float* __restrict__ q, const float* __restrict__ k,
                        const int* __restrict__ topidx, float* __restrict__ scores, int U) {
  int bh = blockIdx.y;
  int b = bh >> 3, h = bh & 7;
  __shared__ _Float16 sq[UPAD * DD];

  int tid = threadIdx.x;
  for (int i = tid; i < UPAD * DD; i += 128) {
    int u = i >> 6, d = i & 63;
    float val = 0.0f;
    if (u < U) {
      int l = topidx[bh * UPAD + u];
      val = q[(((long)b * LL + l) * HH + h) * DD + d];
    }
    sq[i] = (_Float16)val;
  }
  __syncthreads();

  int lane = tid & 31, wave = tid >> 5;
  int half = lane >> 4;

  v16h afr[3][2];
#pragma unroll
  for (int mt = 0; mt < 3; ++mt)
#pragma unroll
    for (int kc = 0; kc < 2; ++kc)
      afr[mt][kc] = load_a_qred(sq, mt, kc, lane);

  const float scale = 0.125f;   // 1/sqrt(64)
#pragma unroll
  for (int t = 0; t < 4; ++t) {
    int nbase = blockIdx.x * 256 + (wave * 4 + t) * 16;
    v16h bf0 = load_b_k(k, b, h, nbase, 0, lane);
    v16h bf1 = load_b_k(k, b, h, nbase, 1, lane);
#pragma unroll
    for (int mt = 0; mt < 3; ++mt) {
      v8f c = {};
      c = wmma_f16(afr[mt][0], bf0, c);
      c = wmma_f16(afr[mt][1], bf1, c);
      int n = nbase + (lane & 15);
#pragma unroll
      for (int r = 0; r < 8; ++r) {
        int u = mt * 16 + half * 8 + r;
        scores[((long)bh * UPAD + u) * LL + n] = c[r] * scale;
      }
    }
  }
}

// ---------------------------------------------------------------------------
// Kernel 5: per-row max and sum(exp) over 4096 (rows = 32*UPAD)
// ---------------------------------------------------------------------------
__global__ __launch_bounds__(256)
void rowstats_kernel(const float* __restrict__ scores,
                     float* __restrict__ rowmax, float* __restrict__ rowsum) {
  int row = blockIdx.x;
  const float* s = scores + (long)row * LL;
  __shared__ float red[256];
  int tid = threadIdx.x;

  float mx = -INFINITY;
  for (int i = tid; i < LL; i += 256) mx = fmaxf(mx, s[i]);
  red[tid] = mx;
  __syncthreads();
  for (int k = 128; k > 0; k >>= 1) {
    if (tid < k) red[tid] = fmaxf(red[tid], red[tid + k]);
    __syncthreads();
  }
  float m0 = red[0];
  __syncthreads();

  float sum = 0.0f;
  for (int i = tid; i < LL; i += 256) sum += __expf(s[i] - m0);
  red[tid] = sum;
  __syncthreads();
  for (int k = 128; k > 0; k >>= 1) {
    if (tid < k) red[tid] += red[tid + k];
    __syncthreads();
  }
  if (tid == 0) { rowmax[row] = m0; rowsum[row] = red[0]; }
}

// ---------------------------------------------------------------------------
// Kernel 6: update[u,d] = P(48x4096) @ v(4096x64), scatter rows u<U to out
// one block per (b,h), 4 waves: wave -> 16-wide d tile, 3 M-tiles, 128 K steps
// ---------------------------------------------------------------------------
__global__ __launch_bounds__(128)
void update_wmma_kernel(const float* __restrict__ v, const int* __restrict__ topidx,
                        const float* __restrict__ scores,
                        const float* __restrict__ rowmax, const float* __restrict__ rowsum,
                        float* __restrict__ out, int U) {
  int bh = blockIdx.x;
  int b = bh >> 3, h = bh & 7;
  __shared__ int   sidx[UPAD];
  __shared__ float smax[UPAD];
  __shared__ float sinv[UPAD];

  int tid = threadIdx.x;
  if (tid < UPAD) {
    sidx[tid] = topidx[bh * UPAD + tid];
    smax[tid] = rowmax[bh * UPAD + tid];
    sinv[tid] = 1.0f / rowsum[bh * UPAD + tid];
  }
  __syncthreads();

  int lane = tid & 31, wave = tid >> 5;
  int half = lane >> 4;
  int dbase = wave * 16;

  v8f acc0 = {}, acc1 = {}, acc2 = {};
  for (int kb = 0; kb < LL / 32; ++kb) {
    v16h bf = load_b_v(v, b, h, kb, dbase, lane);
    v16h a0 = load_a_p(scores, smax, sinv, bh, 0, kb, lane);
    acc0 = wmma_f16(a0, bf, acc0);
    v16h a1 = load_a_p(scores, smax, sinv, bh, 1, kb, lane);
    acc1 = wmma_f16(a1, bf, acc1);
    v16h a2 = load_a_p(scores, smax, sinv, bh, 2, kb, lane);
    acc2 = wmma_f16(a2, bf, acc2);
  }

  int d = dbase + (lane & 15);
#pragma unroll
  for (int mt = 0; mt < 3; ++mt) {
    v8f acc = (mt == 0) ? acc0 : (mt == 1) ? acc1 : acc2;
#pragma unroll
    for (int r = 0; r < 8; ++r) {
      int u = mt * 16 + half * 8 + r;
      if (u < U) {
        int l = sidx[u];
        out[((long)bh * LL + l) * DD + d] = acc[r];
      }
    }
  }
}

// ---------------------------------------------------------------------------
// Host launcher
// ---------------------------------------------------------------------------
extern "C" void kernel_launch(void* const* d_in, const int* in_sizes, int n_in,
                              void* d_out, int out_size, void* d_ws, size_t ws_size,
                              hipStream_t stream) {
  const float* q   = (const float*)d_in[0];
  const float* k   = (const float*)d_in[1];
  const float* v   = (const float*)d_in[2];
  const int*   idx = (const int*)d_in[3];
  float* out = (float*)d_out;
  int U = in_sizes[3] / LL;   // = 45

  char* ws = (char*)d_ws;
  float* mvals  = (float*)(ws);                           // 32*4096 f32    = 512 KB
  int*   topidx = (int*)  (ws + 524288);                  // 32*48 i32      =   6 KB
  float* vmean  = (float*)(ws + 524288 + 6144);           // 32*64 f32      =   8 KB
  float* rmax   = (float*)(ws + 524288 + 6144 + 8192);    // 32*48 f32
  float* rsum   = (float*)(ws + 524288 + 6144 + 8192 + 6144);
  float* scores = (float*)(ws + 524288 + 6144 + 8192 + 6144 + 6144); // 32*48*4096 f32 = 24 MB

  compute_m_kernel<<<(BB * HH * LL) / 256, 256, 0, stream>>>(q, k, idx, mvals, U);
  topk_kernel<<<BB * HH, 256, 0, stream>>>(mvals, topidx, U);
  vmean_kernel<<<BB * HH, 256, 0, stream>>>(v, vmean);
  fill_kernel<<<(BB * HH * LL * DD) / 256, 256, 0, stream>>>(vmean, out);
  scores_wmma_kernel<<<dim3(LL / 256, BB * HH), 128, 0, stream>>>(q, k, topidx, scores, U);
  rowstats_kernel<<<BB * HH * UPAD, 256, 0, stream>>>(scores, rmax, rsum);
  update_wmma_kernel<<<BB * HH, 128, 0, stream>>>(v, topidx, scores, rmax, rsum, out, U);
}
